// BiMambaBlock_5858335392062
// MI455X (gfx1250) — matrix operand order repaired and checked by reading, not compiled, not run on hardware
//
#include <hip/hip_runtime.h>
#include <hip/hip_bf16.h>

// ---------------- problem constants (from reference) ----------------
#define D_MODEL 1024
#define D_STATE 16
#define D_CONVK 4
#define D_INNER 2048
#define DT_RANK 64
#define NPROJ   96          // DT_RANK + 2*D_STATE
#define BATCH   2
#define SEQ     2048
#define NROW    (BATCH*SEQ) // 4096 token rows
#define EPSV    1e-5f
#define CHUNK   128
#define NCHUNK  (SEQ/CHUNK) // 16 scan chunks -> 16x scan parallelism

typedef _Float16 half_t;
typedef __attribute__((ext_vector_type(16))) _Float16 v16h;
typedef __attribute__((ext_vector_type(8)))  _Float16 v8h;   // 16B vector load
typedef __attribute__((ext_vector_type(8)))  float    v8f;

// ---------------- small helpers ----------------
__global__ void k_f32_to_h(const float* __restrict__ in, half_t* __restrict__ out, size_t n) {
  size_t i = (size_t)blockIdx.x * blockDim.x + threadIdx.x;
  if (i < n) out[i] = (half_t)in[i];
}

__global__ void k_copy(const float* __restrict__ in, float* __restrict__ out, size_t n) {
  size_t i = (size_t)blockIdx.x * blockDim.x + threadIdx.x;
  if (i < n) out[i] = in[i];
}

// ---------------- layernorm: f32 in -> f16 out (feeds WMMA in_proj) ----------------
__global__ void k_layernorm(const float* __restrict__ x, const float* __restrict__ g,
                            const float* __restrict__ bb, half_t* __restrict__ out) {
  __shared__ float s1[256], s2[256];
  const int row = blockIdx.x;
  const float* xr = x + (size_t)row * D_MODEL;
  float s = 0.f, q = 0.f;
  for (int c = threadIdx.x; c < D_MODEL; c += 256) { float v = xr[c]; s += v; q += v * v; }
  s1[threadIdx.x] = s; s2[threadIdx.x] = q;
  __syncthreads();
  for (int off = 128; off > 0; off >>= 1) {
    if ((int)threadIdx.x < off) { s1[threadIdx.x] += s1[threadIdx.x + off];
                                  s2[threadIdx.x] += s2[threadIdx.x + off]; }
    __syncthreads();
  }
  const float mu  = s1[0] * (1.f / D_MODEL);
  const float var = s2[0] * (1.f / D_MODEL) - mu * mu;
  const float rs  = rsqrtf(var + EPSV);
  half_t* orow = out + (size_t)row * D_MODEL;
  for (int c = threadIdx.x; c < D_MODEL; c += 256)
    orow[c] = (half_t)((xr[c] - mu) * rs * g[c] + bb[c]);
}

// ---------------- WMMA GEMM: C[M,N] = A[M,K] * W[N,K]^T  (f16 in, f32 acc) -----------
// Block = 256 thr = 8 waves; block tile 128(M) x NT*16(N).
// All 8 waves share the same weight tile -> stage B in LDS (double-buffered):
//   * 256 threads copy the NCOLS x 32-half tile (one 16B chunk each)
//   * rows padded to 40 halves (80B) so 16-lane ds_load_b128 fragment reads are
//     bank-conflict-free; 8x less global B traffic (AI ~13 -> ~42 flops/byte)
//   * fragments for all NT tiles are read before the WMMA chain so ds loads
//     overlap the matrix pipe instead of a full s_wait before every wmma.
// EPI: 0 = store, 1 = softplus(acc + bias[n]), 2 = C += acc (residual accumulate).
template<int NT, int EPI, bool WRITE_H>
__global__ void __launch_bounds__(256)
k_gemm_wmma(const half_t* __restrict__ A, int lda,
            const half_t* __restrict__ W,
            const float* __restrict__ bias,
            float* __restrict__ C, half_t* __restrict__ Ch,
            int N, int K) {
  constexpr int NCOLS = NT * 16;            // weight rows per block tile
  constexpr int RS    = 40;                 // padded row stride in halves (80B)
  __shared__ alignas(16) half_t lb[2][NCOLS * RS];

  const int tid  = threadIdx.x;
  const int lane = tid & 31;
  const int wave = tid >> 5;
  const int m0    = blockIdx.y * 128 + wave * 16;
  const int nbase = blockIdx.x * NCOLS;
  const int ln = lane & 15;
  const int hi = lane >> 4;

  // cooperative B-tile copy: NCOLS rows x 4 chunks of 8 halves
  const int cr = tid >> 2;                  // row 0..63
  const int cc = (tid & 3) * 8;             // half offset within the 32-wide k-slice
  const bool do_copy = (cr < NCOLS);
  const half_t* wsrc = W + (size_t)(nbase + cr) * K + cc;
  half_t* ldst = &lb[0][0] + cr * RS + cc;

  // A fragment per ISA layout: row m0+ln, K chunks {hi*8..+7} and {hi*8+16..+23}
  const half_t* arow = A + (size_t)(m0 + ln) * lda + hi * 8;
  // B fragment from LDS: row j*16+ln, K chunks {hi*8..+7} and {hi*8+16..+23}
  const half_t* lrd = &lb[0][0] + (size_t)ln * RS + hi * 8;

  v8f acc[NT];
#pragma unroll
  for (int j = 0; j < NT; ++j) acc[j] = (v8f){};

  if (do_copy) *(v8h*)(ldst) = *(const v8h*)(wsrc);          // stage buffer 0

  for (int k0 = 0; k0 < K; k0 += 32) {
    const int buf = (k0 >> 5) & 1;
    __syncthreads();                                          // staged tile visible
    if (k0 + 32 < K && do_copy)                               // prefetch next k-slice
      *(v8h*)(ldst + (buf ^ 1) * (NCOLS * RS)) = *(const v8h*)(wsrc + k0 + 32);

    union { v16h v; v8h h[2]; } a;
    a.h[0] = *(const v8h*)(arow + k0);
    a.h[1] = *(const v8h*)(arow + k0 + 16);
    __builtin_prefetch(arow + k0 + 32, 0, 3);                 // global_prefetch_b8

    union { v16h v; v8h h[2]; } bf[NT];
    const half_t* lcur = lrd + buf * (NCOLS * RS);
#pragma unroll
    for (int j = 0; j < NT; ++j) {                            // all ds reads first
      bf[j].h[0] = *(const v8h*)(lcur + j * 16 * RS);
      bf[j].h[1] = *(const v8h*)(lcur + j * 16 * RS + 16);
    }
#pragma unroll
    for (int j = 0; j < NT; ++j)                              // then the WMMA chain
      acc[j] = __builtin_amdgcn_wmma_f32_16x16x32_f16(
                   false, a.v, false, bf[j].v, (short)0, acc[j], false, false);
  }

#pragma unroll
  for (int j = 0; j < NT; ++j) {
    const int n = nbase + j * 16 + ln;
    const float bv = (EPI == 1) ? bias[n] : 0.f;
#pragma unroll
    for (int r = 0; r < 8; ++r) {          // C/D layout: VGPR r -> M = m0 + r + 8*hi
      const int m = m0 + r + hi * 8;
      float v = acc[j][r];
      const size_t idx = (size_t)m * N + n;
      if (EPI == 1) {                      // softplus(x + b)
        v += bv;
        v = (v > 20.f) ? v : __logf(1.f + __expf(v));
        C[idx] = v;
      } else if (EPI == 2) {
        C[idx] += v;                       // residual accumulate into d_out
      } else {
        C[idx] = v;
      }
      if (WRITE_H) Ch[idx] = (half_t)v;
    }
  }
}

// ---------------- depthwise causal conv(4) + SiLU; direction-aware ----------------
// u lives in xz columns [0, D_INNER); threads consecutive in d -> coalesced.
__global__ void k_conv_silu(const float* __restrict__ xz, const float* __restrict__ convw,
                            const float* __restrict__ convb, float* __restrict__ ucf,
                            half_t* __restrict__ uch, int backward) {
  size_t idx = (size_t)blockIdx.x * blockDim.x + threadIdx.x;
  if (idx >= (size_t)NROW * D_INNER) return;
  const int d = idx % D_INNER;
  const int l = (int)((idx / D_INNER) % SEQ);
  const int b = (int)(idx / ((size_t)D_INNER * SEQ));
  float acc = convb[d];
#pragma unroll
  for (int j = 0; j < D_CONVK; ++j) {
    const int lp = backward ? (l + (D_CONVK - 1) - j) : (l - (D_CONVK - 1) + j);
    if (lp >= 0 && lp < SEQ)
      acc += xz[((size_t)b * SEQ + lp) * (2 * D_INNER) + d] * convw[d * D_CONVK + j];
  }
  const float s = acc * (1.f / (1.f + __expf(-acc)));   // SiLU
  ucf[idx] = s;
  uch[idx] = (half_t)s;
}

// ---------------- chunked selective scan, pass A: local scans ----------------
// thread = (b, chunk, d). h_t = exp(dt*A)*h + (dt*u)*B ; y_local = <h, C>.
// Also emits per-chunk cumulative decay product P (for the chunk combine).
__global__ void k_scan_chunk(const float* __restrict__ dt, const float* __restrict__ ucf,
                             const float* __restrict__ xdbl, const float* __restrict__ Alog,
                             float* __restrict__ ybuf, float* __restrict__ hend,
                             float* __restrict__ pprod, int backward) {
  const int d  = blockIdx.x * blockDim.x + threadIdx.x;
  const int b  = blockIdx.y / NCHUNK;
  const int c  = blockIdx.y % NCHUNK;
  float Ar[D_STATE], h[D_STATE], P[D_STATE];
#pragma unroll
  for (int n = 0; n < D_STATE; ++n) {
    Ar[n] = -__expf(Alog[d * D_STATE + n]);
    h[n] = 0.f; P[n] = 1.f;
  }
  for (int i = 0; i < CHUNK; ++i) {
    const int t = c * CHUNK + i;
    const int l = backward ? (SEQ - 1 - t) : t;          // time reversal for bwd branch
    const size_t row = (size_t)b * SEQ + l;
    const float dtv = dt[row * D_INNER + d];
    const float bu  = dtv * ucf[row * D_INNER + d];
    const float* bc = xdbl + row * NPROJ + DT_RANK;      // B then C, 16 each (uniform addr)
    float y = 0.f;
#pragma unroll
    for (int n = 0; n < D_STATE; ++n) {
      const float a = __expf(dtv * Ar[n]);
      h[n] = a * h[n] + bu * bc[n];
      P[n] *= a;
      y += h[n] * bc[D_STATE + n];
    }
    ybuf[row * D_INNER + d] = y;
  }
  const size_t base = (((size_t)b * NCHUNK + c) * D_INNER + d) * D_STATE;
#pragma unroll
  for (int n = 0; n < D_STATE; ++n) { hend[base + n] = h[n]; pprod[base + n] = P[n]; }
}

// ---------------- pass B: sequential combine of 16 chunk states (tiny) ----------------
__global__ void k_scan_combine(const float* __restrict__ hend, const float* __restrict__ pprod,
                               float* __restrict__ hinit) {
  const int idx = blockIdx.x * blockDim.x + threadIdx.x;
  if (idx >= BATCH * D_INNER) return;
  const int b = idx / D_INNER, d = idx % D_INNER;
  float h[D_STATE];
#pragma unroll
  for (int n = 0; n < D_STATE; ++n) h[n] = 0.f;
  for (int c = 0; c < NCHUNK; ++c) {
    const size_t base = (((size_t)b * NCHUNK + c) * D_INNER + d) * D_STATE;
#pragma unroll
    for (int n = 0; n < D_STATE; ++n) {
      hinit[base + n] = h[n];                                // state entering chunk c
      h[n] = pprod[base + n] * h[n] + hend[base + n];
    }
  }
}

// ---------------- pass C: cross-chunk correction + fused gating -> f16 for out_proj ----
// y_t += <Acum_t * h_init, C_t>;  y = (y + u*D) * silu(z);  store f16.
__global__ void k_scan_fixup(const float* __restrict__ dt, const float* __restrict__ ucf,
                             const float* __restrict__ xz, const float* __restrict__ xdbl,
                             const float* __restrict__ Alog, const float* __restrict__ Dp,
                             const float* __restrict__ hinit, const float* __restrict__ ybuf,
                             half_t* __restrict__ yh, int backward) {
  const int d = blockIdx.x * blockDim.x + threadIdx.x;
  const int b = blockIdx.y / NCHUNK;
  const int c = blockIdx.y % NCHUNK;
  const bool have = (c > 0);                               // chunk 0 has h_init == 0
  float Ar[D_STATE], Acum[D_STATE], h0[D_STATE];
  const size_t base = (((size_t)b * NCHUNK + c) * D_INNER + d) * D_STATE;
#pragma unroll
  for (int n = 0; n < D_STATE; ++n) {
    Ar[n] = -__expf(Alog[d * D_STATE + n]);
    Acum[n] = 1.f;
    h0[n] = hinit[base + n];
  }
  const float dv = Dp[d];
  for (int i = 0; i < CHUNK; ++i) {
    const int t = c * CHUNK + i;
    const int l = backward ? (SEQ - 1 - t) : t;
    const size_t row = (size_t)b * SEQ + l;
    float yv = ybuf[row * D_INNER + d];
    if (have) {
      const float dtv = dt[row * D_INNER + d];
      const float* cv = xdbl + row * NPROJ + DT_RANK + D_STATE;
      float corr = 0.f;
#pragma unroll
      for (int n = 0; n < D_STATE; ++n) {
        Acum[n] *= __expf(dtv * Ar[n]);
        corr += Acum[n] * h0[n] * cv[n];
      }
      yv += corr;
    }
    const float uv = ucf[row * D_INNER + d];
    const float zv = xz[row * (2 * D_INNER) + D_INNER + d];
    const float sz = zv * (1.f / (1.f + __expf(-zv)));     // silu(z)
    yv = (yv + uv * dv) * sz;
    yh[row * D_INNER + d] = (half_t)yv;
  }
}

// ------------------------------- host orchestration -------------------------------
extern "C" void kernel_launch(void* const* d_in, const int* in_sizes, int n_in,
                              void* d_out, int out_size, void* d_ws, size_t ws_size,
                              hipStream_t stream) {
  (void)in_sizes; (void)n_in; (void)out_size; (void)ws_size;

  const float* x  = (const float*)d_in[0];
  const float* ng = (const float*)d_in[1];
  const float* nb = (const float*)d_in[2];

  // bump allocator over d_ws (~240 MB total)
  char* ws = (char*)d_ws;
  size_t off = 0;
  auto alloc = [&](size_t bytes) -> void* {
    void* p = ws + off;
    off = (off + bytes + 255) & ~(size_t)255;
    return p;
  };
  half_t* xnh   = (half_t*)alloc((size_t)NROW * D_MODEL * 2);
  half_t* Win_h = (half_t*)alloc((size_t)2 * D_INNER * D_MODEL * 2);
  half_t* Wx_h  = (half_t*)alloc((size_t)NPROJ * D_INNER * 2);
  half_t* Wdt_h = (half_t*)alloc((size_t)D_INNER * DT_RANK * 2);
  half_t* Wout_h= (half_t*)alloc((size_t)D_MODEL * D_INNER * 2);
  float*  xz    = (float*)alloc((size_t)NROW * 2 * D_INNER * 4);   // [u | z]
  float*  ucf   = (float*)alloc((size_t)NROW * D_INNER * 4);
  half_t* uch   = (half_t*)alloc((size_t)NROW * D_INNER * 2);
  float*  xdbl  = (float*)alloc((size_t)NROW * NPROJ * 4);
  half_t* xdblh = (half_t*)alloc((size_t)NROW * NPROJ * 2);
  float*  dtb   = (float*)alloc((size_t)NROW * D_INNER * 4);
  float*  ybuf  = (float*)alloc((size_t)NROW * D_INNER * 4);
  half_t* yh    = (half_t*)alloc((size_t)NROW * D_INNER * 2);
  const size_t hsz = (size_t)BATCH * NCHUNK * D_INNER * D_STATE;
  float*  hend  = (float*)alloc(hsz * 4);
  float*  pprod = (float*)alloc(hsz * 4);
  float*  hinit = (float*)alloc(hsz * 4);

  // shared prologue: layernorm -> f16, d_out = x (residual base)
  k_layernorm<<<NROW, 256, 0, stream>>>(x, ng, nb, xnh);
  { size_t n = (size_t)NROW * D_MODEL;
    k_copy<<<(unsigned)((n + 255) / 256), 256, 0, stream>>>(x, (float*)d_out, n); }

  for (int dir = 0; dir < 2; ++dir) {          // 0 = fwd, 1 = time-reversed branch
    const int o = 3 + dir * 9;
    const float* Win   = (const float*)d_in[o + 0];
    const float* convw = (const float*)d_in[o + 1];
    const float* convb = (const float*)d_in[o + 2];
    const float* Wx    = (const float*)d_in[o + 3];
    const float* Wdt   = (const float*)d_in[o + 4];
    const float* bdt   = (const float*)d_in[o + 5];
    const float* Alog  = (const float*)d_in[o + 6];
    const float* Dp    = (const float*)d_in[o + 7];
    const float* Wout  = (const float*)d_in[o + 8];

    // weight conversion f32 -> f16 (deterministic every launch; graph-safe)
    { size_t n = (size_t)2 * D_INNER * D_MODEL;
      k_f32_to_h<<<(unsigned)((n + 255) / 256), 256, 0, stream>>>(Win, Win_h, n); }
    { size_t n = (size_t)NPROJ * D_INNER;
      k_f32_to_h<<<(unsigned)((n + 255) / 256), 256, 0, stream>>>(Wx, Wx_h, n); }
    { size_t n = (size_t)D_INNER * DT_RANK;
      k_f32_to_h<<<(unsigned)((n + 255) / 256), 256, 0, stream>>>(Wdt, Wdt_h, n); }
    { size_t n = (size_t)D_MODEL * D_INNER;
      k_f32_to_h<<<(unsigned)((n + 255) / 256), 256, 0, stream>>>(Wout, Wout_h, n); }

    // in_proj: xz = xn * Win^T   [4096 x 4096]   NT=4 -> 64-wide block tiles
    k_gemm_wmma<4, 0, false><<<dim3((2 * D_INNER) / 64, NROW / 128), 256, 0, stream>>>(
        xnh, D_MODEL, Win_h, nullptr, xz, nullptr, 2 * D_INNER, D_MODEL);

    // depthwise conv + SiLU (direction-aware causality)
    { size_t n = (size_t)NROW * D_INNER;
      k_conv_silu<<<(unsigned)((n + 255) / 256), 256, 0, stream>>>(
          xz, convw, convb, ucf, uch, dir); }

    // x_proj: xdbl = uc * Wx^T   [4096 x 96]   NT=2 -> grid.x = 3 covers N exactly
    k_gemm_wmma<2, 0, true><<<dim3(NPROJ / 32, NROW / 128), 256, 0, stream>>>(
        uch, D_INNER, Wx_h, nullptr, xdbl, xdblh, NPROJ, D_INNER);

    // dt_proj + bias + softplus   [4096 x 2048], K=64, A strided inside xdblh
    k_gemm_wmma<4, 1, false><<<dim3(D_INNER / 64, NROW / 128), 256, 0, stream>>>(
        xdblh, NPROJ, Wdt_h, bdt, dtb, nullptr, D_INNER, DT_RANK);

    // chunked selective scan (3 passes) + fused gating
    k_scan_chunk<<<dim3(D_INNER / 256, BATCH * NCHUNK), 256, 0, stream>>>(
        dtb, ucf, xdbl, Alog, ybuf, hend, pprod, dir);
    k_scan_combine<<<(BATCH * D_INNER + 255) / 256, 256, 0, stream>>>(hend, pprod, hinit);
    k_scan_fixup<<<dim3(D_INNER / 256, BATCH * NCHUNK), 256, 0, stream>>>(
        dtb, ucf, xz, xdbl, Alog, Dp, hinit, ybuf, yh, dir);

    // out_proj accumulated into residual: d_out += y * Wout^T
    k_gemm_wmma<4, 2, false><<<dim3(D_MODEL / 64, NROW / 128), 256, 0, stream>>>(
        yh, D_INNER, Wout_h, nullptr, (float*)d_out, nullptr, D_MODEL, D_INNER);
  }
}
